// MultiHeadAttention_51814485458985
// MI455X (gfx1250) — matrix operand order repaired and checked by reading, not compiled, not run on hardware
//
#include <hip/hip_runtime.h>
#include <hip/hip_bf16.h>
#include <cstdint>

typedef __attribute__((ext_vector_type(16))) _Float16 v16h;
typedef __attribute__((ext_vector_type(8)))  _Float16 v8h;
typedef __attribute__((ext_vector_type(8)))  float    v8f;
typedef __attribute__((ext_vector_type(4)))  float    v4f;
typedef __attribute__((ext_vector_type(4)))  unsigned int u32x4;
typedef __attribute__((ext_vector_type(8)))  int      i32x8;
typedef __attribute__((ext_vector_type(4)))  int      i32x4;

#if defined(__has_builtin)
#if __has_builtin(__builtin_amdgcn_tensor_load_to_lds) && \
    __has_builtin(__builtin_amdgcn_s_wait_tensorcnt)
#define HAVE_TDM 1
#endif
#endif
#ifndef HAVE_TDM
#define HAVE_TDM 0
#endif

#define DEVI static __device__ __forceinline__

// Assemble a v16h fragment from two contiguous 16-byte chunks.
DEVI v16h load2x8(const _Float16* p0, const _Float16* p1) {
  v8h lo = *(const v8h*)p0;
  v8h hi = *(const v8h*)p1;
  v16h r;
#pragma unroll
  for (int i = 0; i < 8; ++i) { r[i] = lo[i]; r[i + 8] = hi[i]; }
  return r;
}

DEVI v8f wmma_f16(v16h a, v16h b, v8f c) {
  return __builtin_amdgcn_wmma_f32_16x16x32_f16(false, a, false, b, (short)0, c,
                                                false, false);
}

// ---------------------------------------------------------------------------
// GEMM with bias:  C[m,n] = sum_k A[m,k] * W[n,k] + bias[n]
// MODE 0: A = fp32 x  -> scatter f16 into Q/K/V [B,H,T,hd]   (N = 2304)
// MODE 1: A = f16  y  -> fp32 output, row-major [M,N]         (N = 768)
// 128 threads (4 waves); tile 64(M) x 64(N), K-step 32.
// ---------------------------------------------------------------------------
template <int MODE, int KDIM, int NDIM>
__global__ __launch_bounds__(128) void gemm_wmma(
    const void* __restrict__ Ain, const float* __restrict__ W,
    const float* __restrict__ bias, _Float16* __restrict__ qh,
    _Float16* __restrict__ kh, _Float16* __restrict__ vh,
    float* __restrict__ outf) {
  constexpr int AS = 40;  // 80B row stride: odd multiple of 16B, b128-clean
  __shared__ __align__(16) _Float16 As[64][AS];
  __shared__ __align__(16) _Float16 Bs[64][AS];

  const int n0 = blockIdx.x * 64;
  const int m0 = blockIdx.y * 64;
  const int tid = threadIdx.x;
  const int wave = tid >> 5, lane = tid & 31;
  const int lane16 = lane & 15, hi = lane >> 4;
  const int abase = 8 * hi;
  const int bbase = 16 * hi;

  const int lrow = tid >> 1;
  const int lcol = (tid & 1) * 16;

  v8f acc[4] = {};

#pragma unroll 1
  for (int k0 = 0; k0 < KDIM; k0 += 32) {
    // ---- stage A tile (64x32), vector LDS stores ----
    if (MODE == 0) {
      const v4f* apv =
          (const v4f*)((const float*)Ain + (size_t)(m0 + lrow) * KDIM + k0 + lcol);
      v8h h0, h1;
#pragma unroll
      for (int j = 0; j < 2; ++j) {
        v4f t = apv[j];
#pragma unroll
        for (int e = 0; e < 4; ++e) h0[j * 4 + e] = (_Float16)t[e];
      }
#pragma unroll
      for (int j = 0; j < 2; ++j) {
        v4f t = apv[2 + j];
#pragma unroll
        for (int e = 0; e < 4; ++e) h1[j * 4 + e] = (_Float16)t[e];
      }
      *(v8h*)&As[lrow][lcol] = h0;
      *(v8h*)&As[lrow][lcol + 8] = h1;
    } else {
      const v8h* apv =
          (const v8h*)((const _Float16*)Ain + (size_t)(m0 + lrow) * KDIM + k0 + lcol);
      *(v8h*)&As[lrow][lcol] = apv[0];
      *(v8h*)&As[lrow][lcol + 8] = apv[1];
    }
    // ---- stage B tile: Bs[n][k] = W[n0+n][k0+k] ----
    {
      const v4f* wpv = (const v4f*)(W + (size_t)(n0 + lrow) * KDIM + k0 + lcol);
      v8h h0, h1;
#pragma unroll
      for (int j = 0; j < 2; ++j) {
        v4f t = wpv[j];
#pragma unroll
        for (int e = 0; e < 4; ++e) h0[j * 4 + e] = (_Float16)t[e];
      }
#pragma unroll
      for (int j = 0; j < 2; ++j) {
        v4f t = wpv[2 + j];
#pragma unroll
        for (int e = 0; e < 4; ++e) h1[j * 4 + e] = (_Float16)t[e];
      }
      *(v8h*)&Bs[lrow][lcol] = h0;
      *(v8h*)&Bs[lrow][lcol + 8] = h1;
    }
    __syncthreads();

    v16h afrag = load2x8(&As[wave * 16 + lane16][abase],
                         &As[wave * 16 + lane16][abase + 16]);
#pragma unroll
    for (int nt = 0; nt < 4; ++nt) {
      v16h bfrag = load2x8(&Bs[nt * 16 + lane16][abase],
                           &Bs[nt * 16 + lane16][abase + 16]);
      acc[nt] = wmma_f16(afrag, bfrag, acc[nt]);
    }
    __syncthreads();
  }

  // ---- epilogue ----
#pragma unroll
  for (int nt = 0; nt < 4; ++nt) {
    const int ng = n0 + nt * 16 + lane16;
    const float bv = bias[ng];
#pragma unroll
    for (int r = 0; r < 8; ++r) {
      const int mg = m0 + wave * 16 + r + 8 * hi;
      float v = acc[nt][r] + bv;
      if (MODE == 0) {
        const int sel = ng / 768, rem = ng % 768;
        const int h = rem >> 6, d = rem & 63;
        const int b = mg >> 11, t = mg & 2047;
        _Float16* dst = (sel == 0) ? qh : (sel == 1) ? kh : vh;
        dst[(((size_t)b * 12 + h) * 2048 + t) * 64 + d] = (_Float16)v;
      } else {
        outf[(size_t)mg * NDIM + ng] = v;
      }
    }
  }
}

// ---------------------------------------------------------------------------
// Flash attention, S^T formulation: one block = one (b,h) x 64 query rows.
//  S^T = K * Q^T  -> each lane owns ONE query column; softmax stats are
//  per-lane reductions + a single shfl_xor(16) to merge the two half-waves.
//  K tile staged by the Tensor Data Mover (async DMA, padded rows) when
//  available; V staged transposed; P repacked via two b128 LDS stores.
// ---------------------------------------------------------------------------
__global__ __launch_bounds__(128) void flash_attn(
    const _Float16* __restrict__ Qh, const _Float16* __restrict__ Kh,
    const _Float16* __restrict__ Vh, _Float16* __restrict__ Yh) {
  constexpr int T = 2048, HD = 64, H = 12;
  __shared__ __align__(16) _Float16 Kt[32][72];     // 144B rows (TDM pad)
  __shared__ __align__(16) _Float16 Vt[64][40];     // [hd][key] transposed
  __shared__ __align__(16) _Float16 Pq[4][16][40];  // per-wave P, [q][key]

  const int bh = blockIdx.y;
  const int q0 = blockIdx.x * 64;
  const int b = bh / H, h = bh % H;
  const size_t base = (size_t)bh * T * HD;

  const int tid = threadIdx.x;
  const int wave = tid >> 5, lane = tid & 31;
  const int lane16 = lane & 15, hi = lane >> 4;
  const int abase = 8 * hi, bbase = 16 * hi;

  // Q^T B-fragments in registers: lane = query column, elements = hd.
  const _Float16* qp = Qh + base + (size_t)(q0 + wave * 16 + lane16) * HD;
  v16h qb[2];
#pragma unroll
  for (int f = 0; f < 2; ++f)
    qb[f] = load2x8(qp + f * 32 + abase, qp + f * 32 + abase + 16);

  const int q_lane = q0 + wave * 16 + lane16;  // this lane's query
  float mS = -3.0e38f, lS = 0.0f;
  v8f O[4] = {};

  const int nkb = q0 / 32 + 2;  // causal: key blocks <= q0+63
  const int krow = tid >> 2;       // 0..31
  const int c0 = (tid & 3) * 16;   // 0,16,32,48

#if HAVE_TDM
  const unsigned int kt_lds = (unsigned int)(uintptr_t)&Kt[0][0];
#endif

#pragma unroll 1
  for (int kb = 0; kb < nkb; ++kb) {
    const int kk = kb * 32;

#if HAVE_TDM
    if (wave == 0) {
      // D# for a 2D f16 tile: 32 rows x 64 elems, row stride 64 elems,
      // LDS pad: +4 dwords every 32 dwords -> 144B LDS row stride.
      const uint64_t ga = (uint64_t)(uintptr_t)(Kh + base + (size_t)kk * HD);
      u32x4 g0;
      g0[0] = 1u;                                   // count=1, no gather
      g0[1] = kt_lds;                               // lds_addr
      g0[2] = (unsigned int)ga;                     // global_addr[31:0]
      g0[3] = (unsigned int)((ga >> 32) & 0x1FFFFFFull) | (2u << 30);  // type=2
      i32x8 g1;
      g1[0] = (1 << 16) | (1 << 20) | (4 << 22) | (3 << 25);
      //        data_size=2B | pad_en | pad_interval=32dw | pad_amount=4dw
      g1[1] = (HD << 16);          // tensor_dim0 = 64 (low bits in [31:16])
      g1[2] = (int)(2048u << 16);  // tensor_dim0 hi=0, tensor_dim1 lo=2048
      g1[3] = (HD << 16);          // tensor_dim1 hi=0, tile_dim0 = 64
      g1[4] = 32;                  // tile_dim1 = 32, tile_dim2 = 0
      g1[5] = HD;                  // tensor_dim0_stride = 64
      g1[6] = 0;
      g1[7] = 0;
      i32x4 gz = {0, 0, 0, 0};
      i32x8 gz8 = {0, 0, 0, 0, 0, 0, 0, 0};
      __builtin_amdgcn_tensor_load_to_lds(g0, g1, gz, gz, gz8, 0);
    }
#else
    {
      const v8h* kpv = (const v8h*)(Kh + base + (size_t)(kk + krow) * HD + c0);
      *(v8h*)&Kt[krow][c0] = kpv[0];
      *(v8h*)&Kt[krow][c0 + 8] = kpv[1];
    }
#endif
    // ---- stage V transposed (overlaps the K DMA) ----
    {
      const _Float16* vp = Vh + base + (size_t)(kk + krow) * HD + c0;
      v8h u0 = ((const v8h*)vp)[0], u1 = ((const v8h*)vp)[1];
#pragma unroll
      for (int e = 0; e < 8; ++e) {
        Vt[c0 + e][krow] = u0[e];
        Vt[c0 + 8 + e][krow] = u1[e];
      }
    }
#if HAVE_TDM
    if (wave == 0) __builtin_amdgcn_s_wait_tensorcnt(0);
#endif
    __syncthreads();

    // ---- S^T = K * Q^T : rows = keys (VGPRs), lanes = queries ----
    v8f st[2];
#pragma unroll
    for (int nt = 0; nt < 2; ++nt) {
      v8f c = {};
#pragma unroll
      for (int f = 0; f < 2; ++f) {
        v16h ka = load2x8(&Kt[nt * 16 + lane16][f * 32 + abase],
                          &Kt[nt * 16 + lane16][f * 32 + abase + 16]);
        c = wmma_f16(ka, qb[f], c);
      }
      st[nt] = c;
    }

    // ---- scale + causal mask (key index now lives in the VGPR dim) ----
#pragma unroll
    for (int nt = 0; nt < 2; ++nt) {
#pragma unroll
      for (int r = 0; r < 8; ++r) {
        const int key = kk + nt * 16 + r + 8 * hi;
        const float v = st[nt][r] * 0.125f;  // 1/sqrt(64)
        st[nt][r] = (key <= q_lane) ? v : -3.0e38f;
      }
    }

    // ---- online softmax: per-lane over 16 keys + one xor-16 merge ----
    float bm = -3.0e38f;
#pragma unroll
    for (int r = 0; r < 8; ++r) bm = fmaxf(bm, fmaxf(st[0][r], st[1][r]));
    bm = fmaxf(bm, __shfl_xor(bm, 16, 32));
    const float mn = fmaxf(mS, bm);
    const float sc = __expf(mS - mn);
    mS = mn;

    float rs = 0.0f;
    v8h p0h, p1h;
#pragma unroll
    for (int r = 0; r < 8; ++r) {
      const float p0 = __expf(st[0][r] - mn);
      const float p1 = __expf(st[1][r] - mn);
      rs += p0 + p1;
      p0h[r] = (_Float16)p0;
      p1h[r] = (_Float16)p1;
    }
    rs += __shfl_xor(rs, 16, 32);
    lS = lS * sc + rs;

    // P stored row-major [q][key]: two b128 stores per lane.
    *(v8h*)&Pq[wave][lane16][8 * hi] = p0h;
    *(v8h*)&Pq[wave][lane16][16 + 8 * hi] = p1h;

    // Rescale O rows (row = query r + 8*hi); broadcast sc from owner lanes.
#pragma unroll
    for (int r = 0; r < 8; ++r) {
      const float scr = __shfl(sc, r + 8 * hi, 32);
#pragma unroll
      for (int dt = 0; dt < 4; ++dt) O[dt][r] *= scr;
    }
    __syncthreads();

    // ---- O += P * V ----
    v16h pfrag = load2x8(&Pq[wave][lane16][abase], &Pq[wave][lane16][abase + 16]);
#pragma unroll
    for (int dt = 0; dt < 4; ++dt) {
      v16h vfrag = load2x8(&Vt[dt * 16 + lane16][abase],
                           &Vt[dt * 16 + lane16][abase + 16]);
      O[dt] = wmma_f16(pfrag, vfrag, O[dt]);
    }
    __syncthreads();
  }

  // ---- normalize and store Y (f16, [B,T,C]) ----
  const float inv = 1.0f / lS;
#pragma unroll
  for (int r = 0; r < 8; ++r) {
    const float invr = __shfl(inv, r + 8 * hi, 32);
    const int qi = q0 + wave * 16 + r + 8 * hi;
#pragma unroll
    for (int dt = 0; dt < 4; ++dt) {
      const int d = h * 64 + dt * 16 + lane16;
      Yh[((size_t)(b * T + qi)) * 768 + d] = (_Float16)(O[dt][r] * invr);
    }
  }
}

// ---------------------------------------------------------------------------
extern "C" void kernel_launch(void* const* d_in, const int* in_sizes, int n_in,
                              void* d_out, int out_size, void* d_ws,
                              size_t ws_size, hipStream_t stream) {
  (void)in_sizes; (void)n_in; (void)out_size; (void)ws_size;
  const float* x = (const float*)d_in[0];       // [4,2048,768]
  const float* W_attn = (const float*)d_in[1];  // [2304,768]
  const float* b_attn = (const float*)d_in[2];  // [2304]
  const float* W_o = (const float*)d_in[3];     // [768,768]
  const float* b_o = (const float*)d_in[4];     // [768]
  float* out = (float*)d_out;                   // [4,2048,768] fp32

  char* ws = (char*)d_ws;
  const size_t seg = (size_t)48 * 2048 * 64 * sizeof(_Float16);  // 12.58 MB
  _Float16* qh = (_Float16*)(ws);
  _Float16* kh = (_Float16*)(ws + seg);
  _Float16* vh = (_Float16*)(ws + 2 * seg);
  _Float16* yh = (_Float16*)(ws + 3 * seg);

  gemm_wmma<0, 768, 2304><<<dim3(2304 / 64, 8192 / 64), 128, 0, stream>>>(
      (const void*)x, W_attn, b_attn, qh, kh, vh, nullptr);

  flash_attn<<<dim3(2048 / 64, 48), 128, 0, stream>>>(qh, kh, vh, yh);

  gemm_wmma<1, 768, 768><<<dim3(768 / 64, 8192 / 64), 128, 0, stream>>>(
      (const void*)yh, W_o, b_o, nullptr, nullptr, nullptr, out);
}
